// ImprovedGNN_63239098466368
// MI455X (gfx1250) — compile-verified, matrix operand order checked by reading, not run on hardware
//
#include <hip/hip_runtime.h>
#include <hip/hip_bf16.h>
#include <math.h>

// ---------------- problem constants (match reference) ----------------
#define NN    30000
#define EE    480000
#define ETOT  (EE + NN)       // with self loops: 510000
#define DIN   128
#define DE    16
#define HH    4
#define DD    64
#define HD    256             // HH*DD
#define GG    64
#define DOUT  256
#define NC    2
#define EPSV  1e-5f
#define SLOPE 0.2f
#define BE    8               // edges per block in score kernel (510000 % 8 == 0)

typedef __attribute__((ext_vector_type(16))) _Float16 v16h;
typedef __attribute__((ext_vector_type(8)))  float    v8f;
typedef unsigned int __attribute__((ext_vector_type(4))) u32x4;
typedef int __attribute__((ext_vector_type(4))) i32x4;
typedef int __attribute__((ext_vector_type(8))) i32x8;

// ---------------- small helpers ----------------
__device__ __forceinline__ void atomicMaxF(float* addr, float val) {
    int old = __float_as_int(*addr);
    while (__int_as_float(old) < val) {
        int assumed = old;
        old = atomicCAS((int*)addr, assumed, __float_as_int(val));
        if (old == assumed) break;
    }
}

__global__ void k_fill(float* __restrict__ p, float v, int n) {
    int i = blockIdx.x * blockDim.x + threadIdx.x;
    if (i < n) p[i] = v;
}

// ---------------- self-loop attr = mean of incoming edge_attr ----------------
__global__ void k_edge_accum(const int* __restrict__ ei, const float* __restrict__ eattr,
                             float* __restrict__ lattr, float* __restrict__ cnt) {
    int t = blockIdx.x * blockDim.x + threadIdx.x;
    if (t >= EE * DE) return;
    int e = t >> 4, k = t & 15;
    int de = ei[EE + e];
    atomicAdd(&lattr[de * DE + k], eattr[t]);
    if (k == 0) atomicAdd(&cnt[de], 1.0f);
}

__global__ void k_loop_div(float* __restrict__ lattr, const float* __restrict__ cnt) {
    int t = blockIdx.x * blockDim.x + threadIdx.x;
    if (t >= NN * DE) return;
    lattr[t] /= fmaxf(cnt[t >> 4], 1.0f);
}

// ---------------- WMMA GEMM: Y[M x 256] = X[M x K] @ W[K x 256] + bias ----------------
// Block = 128 threads (4 waves). Block tile: 16 rows x 128 cols.
// Wave w computes cols [w*32, w*32+32) as two 16x16 WMMA accumulators.
// B K-chunks (64x128 f32) staged to LDS via global_load_async_to_lds_b128 (ASYNCcnt);
// A (16x64) converted to f16 in LDS. K must be a multiple of 64, M of 16.
__global__ __launch_bounds__(128)
void k_gemm_wmma(const float* __restrict__ X, const float* __restrict__ W,
                 const float* __restrict__ bias, float* __restrict__ Y,
                 int K, int do_relu) {
    __shared__ float    Bs[64 * 128];     // 32 KB, raw f32 via async copy
    __shared__ _Float16 As[16 * 64];      // 2 KB, f32->f16 converted

    const int t    = threadIdx.x;          // 0..127
    const int w    = t >> 5;               // wave id 0..3
    const int lane = t & 31;
    const int half = lane >> 4;            // 0: lanes 0-15, 1: lanes 16-31
    const int l    = lane & 15;
    const int rowBase = blockIdx.x * 16;
    const int colBase = blockIdx.y * 128;

    v8f acc0 = {}, acc1 = {};

    for (int k0 = 0; k0 < K; k0 += 64) {
        // ---- async-stage B chunk: 64 rows x 128 cols of f32 (16B per lane-op) ----
#pragma unroll 4
        for (int j = 0; j < 16; ++j) {
            int q    = t + 128 * j;        // vec4 index 0..2047
            int i4   = q * 4;              // f32 index
            int krow = i4 >> 7;            // /128
            int col  = i4 & 127;
            const float* gp = W + (size_t)(k0 + krow) * HD + colBase + col;
            unsigned long long ga = (unsigned long long)(uintptr_t)gp;
            unsigned int la = (unsigned int)(uintptr_t)&Bs[krow * 128 + col];
            asm volatile("global_load_async_to_lds_b128 %0, %1, off"
                         :: "v"(la), "v"(ga) : "memory");
        }
        // ---- stage A tile: 16 rows x 64 k, converted to f16 ----
#pragma unroll
        for (int j = 0; j < 8; ++j) {
            int i = t + 128 * j;           // 0..1023
            int r = i >> 6, k = i & 63;
            As[i] = (_Float16)X[(size_t)(rowBase + r) * K + k0 + k];
        }
        if (k0 + 64 < K)
            __builtin_prefetch(&X[(size_t)(rowBase + l) * K + k0 + 64], 0, 1);

        asm volatile("s_wait_asynccnt 0x0" ::: "memory");
        __syncthreads();

        // ---- 2 K-subchunks of 32, 2 n-subtiles each: 4 WMMAs ----
#pragma unroll
        for (int kk = 0; kk < 64; kk += 32) {
            v16h a, b0, b1;
#pragma unroll
            for (int j = 0; j < 16; ++j) {
                // 16-bit A 16x32 K packing: lanes0-15 -> K {0..7,16..23}; lanes16-31 -> {8..15,24..31}
                int kidx = kk + j + ((j < 8) ? 0 : 8) + half * 8;
                a[j]  = As[l * 64 + kidx];
                b0[j] = (_Float16)Bs[kidx * 128 + (w * 32 + l)];
                b1[j] = (_Float16)Bs[kidx * 128 + (w * 32 + 16 + l)];
            }
            acc0 = __builtin_amdgcn_wmma_f32_16x16x32_f16(false, a, false, b0,
                                                          (short)0, acc0, false, false);
            acc1 = __builtin_amdgcn_wmma_f32_16x16x32_f16(false, a, false, b1,
                                                          (short)0, acc1, false, false);
        }
        __syncthreads();
    }

    // ---- write D: VGPR r holds M=r (lanes 0-15) / M=r+8 (lanes 16-31), N=l ----
    int c0 = colBase + w * 32 + l;
    int c1 = c0 + 16;
    float bv0 = bias[c0], bv1 = bias[c1];
#pragma unroll
    for (int r = 0; r < 8; ++r) {
        int m = rowBase + r + half * 8;
        float v0 = acc0[r] + bv0;
        float v1 = acc1[r] + bv1;
        if (do_relu) { v0 = fmaxf(v0, 0.0f); v1 = fmaxf(v1, 0.0f); }
        Y[(size_t)m * HD + c0] = v0;
        Y[(size_t)m * HD + c1] = v1;
    }
}

// ---------------- attention scores + segment max ----------------
__global__ __launch_bounds__(256)
void k_edge_scores(const float* __restrict__ xs, const float* __restrict__ xd,
                   const int* __restrict__ ei, const float* __restrict__ eattr,
                   const float* __restrict__ lattr, const float* __restrict__ We,
                   const float* __restrict__ att, float* __restrict__ s,
                   float* __restrict__ smax) {
    __shared__ float WeS[DE * HD];   // 16 KB
    __shared__ float attS[HD];
    __shared__ float ea16[DE];
    __shared__ float red[HD];
    const int t = threadIdx.x;

    // --- TDM: DMA We (16x256 f32 = one 4096-element row) into LDS (TENSORcnt) ---
    if (t < 32) {
        unsigned int       la = (unsigned int)(uintptr_t)&WeS[0];
        unsigned long long ga = (unsigned long long)(uintptr_t)We;
        u32x4 g0 = { 1u,                                  // count=1 (valid), no gather
                     la,                                  // lds_addr
                     (unsigned int)ga,                    // global_addr[31:0]
                     ((unsigned int)(ga >> 32) & 0x1FFFFFFu) | (2u << 30) }; // addr[56:32] | type=2
        i32x8 g1 = { (int)(2u << 16),        // data_size = 4B
                     (int)(4096u << 16),     // tensor_dim0[15:0] in bits[63:48]
                     (int)(1u << 16),        // tensor_dim0 hi = 0 | tensor_dim1 = 1
                     (int)(4096u << 16),     // tensor_dim1 hi = 0 | tile_dim0 = 4096
                     1,                      // tile_dim1 = 1, tile_dim2 = 0
                     4096,                   // tensor_dim0_stride[31:0]
                     (int)(4096u << 16),     // stride0 hi = 0 | tensor_dim1_stride lo
                     0 };
        i32x4 z4 = { 0, 0, 0, 0 };
#if defined(__clang_major__) && (__clang_major__ >= 23)
        i32x8 z8 = { 0, 0, 0, 0, 0, 0, 0, 0 };
        __builtin_amdgcn_tensor_load_to_lds(g0, g1, z4, z4, z8, 0);
#else
        __builtin_amdgcn_tensor_load_to_lds(g0, g1, z4, z4, 0);
#endif
    }
    __builtin_amdgcn_s_wait_tensorcnt((short)0);
    __syncthreads();
    // defensive overwrite (descriptor packing is spec-derived; keep results exact)
    for (int i = t; i < DE * HD; i += 256) WeS[i] = We[i];
    attS[t] = att[t];
    __syncthreads();

    const int ebase = blockIdx.x * BE;
    for (int sub = 0; sub < BE; ++sub) {
        const int e = ebase + sub;          // uniform across block; ETOT % BE == 0
        int se, de;
        const float* ea;
        if (e < EE) { se = ei[e]; de = ei[EE + e]; ea = eattr + (long)e * DE; }
        else        { se = de = e - EE;            ea = lattr + (long)(e - EE) * DE; }
        if (t < DE) ea16[t] = ea[t];
        __syncthreads();
        float eev = 0.0f;
#pragma unroll
        for (int k = 0; k < DE; ++k) eev += ea16[k] * WeS[k * HD + t];
        float v = xs[se * HD + t] + xd[de * HD + t] + eev;
        v = (v > 0.0f) ? v : SLOPE * v;     // leaky relu
        red[t] = v * attS[t];
        __syncthreads();
        for (int o = 32; o >= 1; o >>= 1) { // reduce within each 64-wide head segment
            if ((t & 63) < o) red[t] += red[t + o];
            __syncthreads();
        }
        if ((t & 63) == 0) {
            int h = t >> 6;
            float sv = red[t];
            s[e * HH + h] = sv;
            atomicMaxF(&smax[de * HH + h], sv);
        }
        __syncthreads();
    }
}

// ---------------- softmax numerator + denominator ----------------
__global__ void k_expsum(const int* __restrict__ ei, const float* __restrict__ s,
                         const float* __restrict__ smax, float* __restrict__ p,
                         float* __restrict__ denom) {
    int t = blockIdx.x * blockDim.x + threadIdx.x;
    if (t >= ETOT * HH) return;
    int e = t >> 2, h = t & 3;
    int de = (e < EE) ? ei[EE + e] : (e - EE);
    float pv = __expf(s[t] - smax[de * HH + h]);
    p[t] = pv;
    atomicAdd(&denom[de * HH + h], pv);
}

// ---------------- weighted scatter of messages ----------------
__global__ __launch_bounds__(256)
void k_scatter(const int* __restrict__ ei, const float* __restrict__ xs,
               const float* __restrict__ p, const float* __restrict__ denom,
               float* __restrict__ hout) {
    int e = blockIdx.x;
    int f = threadIdx.x;
    int h = f >> 6;
    int se, de;
    if (e < EE) { se = ei[e]; de = ei[EE + e]; }
    else        { se = de = e - EE; }
    float alpha = p[e * HH + h] / denom[de * HH + h];
    atomicAdd(&hout[de * HD + f], xs[se * HD + f] * alpha);
}

// ---------------- bias + relu ----------------
__global__ void k_bias_relu(float* __restrict__ hout, const float* __restrict__ b) {
    int i = blockIdx.x * blockDim.x + threadIdx.x;
    if (i >= NN * HD) return;
    hout[i] = fmaxf(hout[i] + b[i & (HD - 1)], 0.0f);
}

// ---------------- batch-norm: stats then normalize ----------------
__global__ __launch_bounds__(256)
void k_bnstats(const float* __restrict__ hout, float* __restrict__ st) {
    __shared__ float r1[256], r2[256];
    int c = blockIdx.x, t = threadIdx.x;
    float s1 = 0.0f, s2 = 0.0f;
    for (int n = t; n < NN; n += 256) {
        float v = hout[n * HD + c];
        s1 += v; s2 += v * v;
    }
    r1[t] = s1; r2[t] = s2;
    __syncthreads();
    for (int o = 128; o >= 1; o >>= 1) {
        if (t < o) { r1[t] += r1[t + o]; r2[t] += r2[t + o]; }
        __syncthreads();
    }
    if (t == 0) { st[c] = r1[0]; st[HD + c] = r2[0]; }
}

__global__ void k_bnorm(const float* __restrict__ hout, const float* __restrict__ st,
                        const float* __restrict__ w, const float* __restrict__ b,
                        float* __restrict__ hin) {
    int i = blockIdx.x * blockDim.x + threadIdx.x;
    if (i >= NN * HD) return;
    int c = i & (HD - 1);
    float mu = st[c] * (1.0f / NN);
    float var = st[HD + c] * (1.0f / NN) - mu * mu;
    hin[i] = (hout[i] - mu) * rsqrtf(var + EPSV) * w[c] + b[c];
}

// ---------------- graph pooling ----------------
__global__ __launch_bounds__(256)
void k_pool(const float* __restrict__ h, const int* __restrict__ batch,
            float* __restrict__ gsum, float* __restrict__ gmax, float* __restrict__ gcnt) {
    int n = blockIdx.x, f = threadIdx.x;
    int g = batch[n];
    float v = h[n * HD + f];
    atomicAdd(&gsum[g * HD + f], v);
    atomicMaxF(&gmax[g * HD + f], v);
    if (f == 0) atomicAdd(&gcnt[g], 1.0f);
}

__global__ void k_assemble(const float* __restrict__ gsum, const float* __restrict__ gmax,
                           const float* __restrict__ gcnt, float* __restrict__ g) {
    int i = blockIdx.x * blockDim.x + threadIdx.x;
    if (i >= GG * 2 * HD) return;
    int gi = i >> 9, f = i & 511;
    g[i] = (f < HD) ? gsum[gi * HD + f] / fmaxf(gcnt[gi], 1.0f)
                    : gmax[gi * HD + (f - HD)];
}

// ---------------- final tiny classifier: out = gh @ W2 + b2 ----------------
__global__ __launch_bounds__(128)
void k_final(const float* __restrict__ gh, const float* __restrict__ W2,
             const float* __restrict__ b2, float* __restrict__ out) {
    int t = threadIdx.x;                  // 128 = GG * NC
    int gi = t >> 1, c = t & 1;
    float acc = b2[c];
    for (int f = 0; f < DOUT; ++f) acc += gh[gi * DOUT + f] * W2[f * NC + c];
    out[t] = acc;
}

// ---------------- host orchestration ----------------
extern "C" void kernel_launch(void* const* d_in, const int* in_sizes, int n_in,
                              void* d_out, int out_size, void* d_ws, size_t ws_size,
                              hipStream_t stream) {
    const float* x     = (const float*)d_in[0];
    const int*   ei    = (const int*)d_in[1];
    const float* eattr = (const float*)d_in[2];
    const int*   batch = (const int*)d_in[3];
    const float *Wl[3], *bl[3], *Wr[3], *br[3], *We[3], *att[3], *bo[3], *bnw[3], *bnb[3];
    for (int l = 0; l < 3; ++l) {
        int b0 = 4 + l * 9;
        Wl[l]  = (const float*)d_in[b0 + 0];
        bl[l]  = (const float*)d_in[b0 + 1];
        Wr[l]  = (const float*)d_in[b0 + 2];
        br[l]  = (const float*)d_in[b0 + 3];
        We[l]  = (const float*)d_in[b0 + 4];
        att[l] = (const float*)d_in[b0 + 5];
        bo[l]  = (const float*)d_in[b0 + 6];
        bnw[l] = (const float*)d_in[b0 + 7];
        bnb[l] = (const float*)d_in[b0 + 8];
    }
    const float* W1  = (const float*)d_in[31];
    const float* b1v = (const float*)d_in[32];
    const float* W2  = (const float*)d_in[33];
    const float* b2v = (const float*)d_in[34];
    float* out = (float*)d_out;

    // workspace carve-up (floats)
    float* ws = (float*)d_ws;
    size_t off = 0;
    float* cnt   = ws + off; off += NN;
    float* lattr = ws + off; off += (size_t)NN * DE;
    float* xs    = ws + off; off += (size_t)NN * HD;
    float* xd    = ws + off; off += (size_t)NN * HD;
    float* hin   = ws + off; off += (size_t)NN * HD;
    float* hout  = ws + off; off += (size_t)NN * HD;
    float* sarr  = ws + off; off += (size_t)ETOT * HH;
    float* parr  = ws + off; off += (size_t)ETOT * HH;
    float* smax  = ws + off; off += (size_t)NN * HH;
    float* denom = ws + off; off += (size_t)NN * HH;
    float* bnst  = ws + off; off += 2 * HD;
    float* gsum  = ws + off; off += (size_t)GG * HD;
    float* gmax  = ws + off; off += (size_t)GG * HD;
    float* gcnt  = ws + off; off += GG;
    float* gcat  = ws + off; off += (size_t)GG * 2 * HD;
    float* gh    = ws + off; off += (size_t)GG * DOUT;
    (void)ws_size; (void)in_sizes; (void)n_in; (void)out_size;

    // ---- self-loop edge attributes (mean of incoming) ----
    hipMemsetAsync(cnt,   0, NN * sizeof(float), stream);
    hipMemsetAsync(lattr, 0, (size_t)NN * DE * sizeof(float), stream);
    k_edge_accum<<<(EE * DE + 255) / 256, 256, 0, stream>>>(ei, eattr, lattr, cnt);
    k_loop_div<<<(NN * DE + 255) / 256, 256, 0, stream>>>(lattr, cnt);

    const float* X = x;
    int K = DIN;
    for (int l = 0; l < 3; ++l) {
        // node transforms via WMMA (block tile 16x128, 4 waves)
        dim3 gg(NN / 16, HD / 128);
        k_gemm_wmma<<<gg, 128, 0, stream>>>(X, Wl[l], bl[l], xs, K, 0);
        k_gemm_wmma<<<gg, 128, 0, stream>>>(X, Wr[l], br[l], xd, K, 0);

        hipMemsetAsync(hout,  0, (size_t)NN * HD * sizeof(float), stream);
        hipMemsetAsync(denom, 0, (size_t)NN * HH * sizeof(float), stream);
        k_fill<<<(NN * HH + 255) / 256, 256, 0, stream>>>(smax, -INFINITY, NN * HH);

        k_edge_scores<<<ETOT / BE, 256, 0, stream>>>(xs, xd, ei, eattr, lattr,
                                                     We[l], att[l], sarr, smax);
        k_expsum<<<((size_t)ETOT * HH + 255) / 256, 256, 0, stream>>>(ei, sarr, smax, parr, denom);
        k_scatter<<<ETOT, 256, 0, stream>>>(ei, xs, parr, denom, hout);

        k_bias_relu<<<(NN * HD + 255) / 256, 256, 0, stream>>>(hout, bo[l]);
        hipMemsetAsync(bnst, 0, 2 * HD * sizeof(float), stream);
        k_bnstats<<<HD, 256, 0, stream>>>(hout, bnst);
        k_bnorm<<<(NN * HD + 255) / 256, 256, 0, stream>>>(hout, bnst, bnw[l], bnb[l], hin);

        X = hin;
        K = HD;
    }

    // ---- readout ----
    hipMemsetAsync(gsum, 0, (size_t)GG * HD * sizeof(float), stream);
    hipMemsetAsync(gcnt, 0, GG * sizeof(float), stream);
    k_fill<<<(GG * HD + 255) / 256, 256, 0, stream>>>(gmax, -INFINITY, GG * HD);
    k_pool<<<NN, 256, 0, stream>>>(hin, batch, gsum, gmax, gcnt);
    k_assemble<<<(GG * 2 * HD + 255) / 256, 256, 0, stream>>>(gsum, gmax, gcnt, gcat);

    // gh = relu(g @ W1 + b1v)  (M=64, K=512, N=256) via WMMA
    dim3 gr(GG / 16, DOUT / 128);
    k_gemm_wmma<<<gr, 128, 0, stream>>>(gcat, W1, b1v, gh, 2 * HD, 1);

    k_final<<<1, 128, 0, stream>>>(gh, W2, b2v, out);
}
